// EfficientMultiHeadAttention_12902081757473
// MI455X (gfx1250) — matrix operand order repaired and checked by reading, not compiled, not run on hardware
//
#include <hip/hip_runtime.h>

#define D_MODEL 1024
#define NHEADS  16
#define DKH     64
#define BATCH   4
#define SEQ     2048
#define MROWS   (BATCH*SEQ)   // 8192

typedef __attribute__((ext_vector_type(16))) __bf16 v16bf;
typedef __attribute__((ext_vector_type(8)))  float  v8f;
typedef __attribute__((ext_vector_type(4))) unsigned int u32x4;
typedef __attribute__((ext_vector_type(8))) int          i32x8;
typedef __attribute__((ext_vector_type(4))) int          i32x4;

union FragAB { v16bf v; uint4 u[2]; };

__device__ __forceinline__ v16bf ld_frag(const __bf16* p0, const __bf16* p1) {
    FragAB f;
    f.u[0] = *reinterpret_cast<const uint4*>(p0);
    f.u[1] = *reinterpret_cast<const uint4*>(p1);
    return f.v;
}

__device__ __forceinline__ v8f zero8() {
    v8f z;
    #pragma unroll
    for (int i = 0; i < 8; ++i) z[i] = 0.0f;
    return z;
}

__device__ __forceinline__ v8f wmma_bf16(v16bf a, v16bf b, v8f c) {
    return __builtin_amdgcn_wmma_f32_16x16x32_bf16(
        /*neg_a=*/false, a, /*neg_b=*/false, b,
        /*c_mod=*/(short)0, c, /*reuse_a=*/false, /*reuse_b=*/false);
}

// LDS aperture: generic shared pointer carries the wave-relative LDS byte
// offset in addr[31:0] (ISA 10.2 aperture rules).
__device__ __forceinline__ unsigned int lds_off32(const void* p) {
    return (unsigned int)(unsigned long long)p;
}

// TDM: DMA a 2-D bf16 tile (rows x width elems) from global into LDS with
// row padding. pad_iv encodes DWORDs-per-row (3 -> 16 dw = 32 bf16,
// 4 -> 32 dw = 64 bf16); pad_amount fixed at 4 DWORDs (8 bf16) so the LDS
// row stride becomes width+8 bf16, matching the fragment-read layouts.
// D# packing per CDNA5 ISA. Tracked by TENSORcnt.
// Toolchain: 6-arg builtin form (g0, g1, g2, g3, g4, cpol).
__device__ __forceinline__ void tdm_load_tile_bf16(
    const void* gaddr, unsigned int lds_byte_off,
    unsigned int rows, unsigned int width_elems,
    unsigned int row_stride_elems, unsigned int tensor_rows,
    unsigned int pad_iv)
{
    unsigned long long ga = (unsigned long long)gaddr;
    u32x4 g0;
    g0[0] = 1u;                                             // count=1 (valid)
    g0[1] = lds_byte_off;                                   // lds_addr
    g0[2] = (unsigned int)(ga & 0xFFFFFFFFu);               // global_addr lo
    g0[3] = (unsigned int)((ga >> 32) & 0x01FFFFFFu)        // global_addr hi
          | (2u << 30);                                     // type=2 (image)
    i32x8 g1;
    g1[0] = (int)((1u << 16)            // data_size = 1 -> 2 bytes
                | (1u << 20)            // pad_enable
                | (pad_iv << 22)        // pad_interval
                | (3u << 25));          // pad_amount = 4 DWORDs (8 bf16)
    g1[1] = (int)(width_elems << 16);   // tensor_dim0[15:0] (bits 63:48)
    g1[2] = (int)((tensor_rows & 0xFFFFu) << 16); // tensor_dim1[15:0] (95:80)
    g1[3] = (int)(width_elems << 16);   // tensor_dim1 hi=0 | tile_dim0 (127:112)
    g1[4] = (int)rows;                  // tile_dim1 (143:128), tile_dim2=0
    g1[5] = (int)row_stride_elems;      // tensor_dim0_stride lo32 (191:160)
    g1[6] = 0;                          // stride0 hi | dim1_stride lo
    g1[7] = 0;
    i32x4 z4;
    z4[0] = 0; z4[1] = 0; z4[2] = 0; z4[3] = 0;
    i32x8 z8;
    #pragma unroll
    for (int i = 0; i < 8; ++i) z8[i] = 0;
    __builtin_amdgcn_tensor_load_to_lds(g0, g1, z4, z4, z8, 0);
}

// ---------------------------------------------------------------------------
// Kernel 1: QKV projection. C[M,N] = X[M,K] @ W[N,K]^T  (fp32 in, bf16 out)
// Output layout: [b*NH + h][t][dk]. grid = (N/64, M/64, 3); block = 128.
// (Stays on manual loads: needs fp32->bf16 conversion in flight.)
// ---------------------------------------------------------------------------
__global__ __launch_bounds__(128)
void qkv_gemm_kernel(const float* __restrict__ X,
                     const float* __restrict__ Wq,
                     const float* __restrict__ Wk,
                     const float* __restrict__ Wv,
                     __bf16* __restrict__ ws)
{
    const float* W = (blockIdx.z == 0) ? Wq : ((blockIdx.z == 1) ? Wk : Wv);
    __bf16* Out = ws + (size_t)blockIdx.z * ((size_t)MROWS * D_MODEL);

    __shared__ __align__(16) __bf16 Alds[64 * 40];
    __shared__ __align__(16) __bf16 Blds[64 * 40];

    const int tid  = threadIdx.x;
    const int wave = tid >> 5, lane = tid & 31;
    const int l15  = lane & 15, hh = lane >> 4;
    const int wm = (wave >> 1) * 32, wn = (wave & 1) * 32;
    const int M0 = blockIdx.y * 64,  N0 = blockIdx.x * 64;

    v8f acc[2][2];
    #pragma unroll
    for (int i = 0; i < 2; ++i)
        #pragma unroll
        for (int j = 0; j < 2; ++j) acc[i][j] = zero8();

    for (int k0 = 0; k0 < D_MODEL; k0 += 32) {
        __syncthreads();
        #pragma unroll
        for (int i = 0; i < 4; ++i) {
            int c   = tid + 128 * i;
            int row = c >> 3;
            int kk  = (c & 7) * 4;
            float4 a = *reinterpret_cast<const float4*>(X + (size_t)(M0 + row) * D_MODEL + k0 + kk);
            float4 b = *reinterpret_cast<const float4*>(W + (size_t)(N0 + row) * D_MODEL + k0 + kk);
            union { __bf16 h[4]; uint2 u; } pa, pb;
            pa.h[0] = (__bf16)a.x; pa.h[1] = (__bf16)a.y; pa.h[2] = (__bf16)a.z; pa.h[3] = (__bf16)a.w;
            pb.h[0] = (__bf16)b.x; pb.h[1] = (__bf16)b.y; pb.h[2] = (__bf16)b.z; pb.h[3] = (__bf16)b.w;
            *reinterpret_cast<uint2*>(&Alds[row * 40 + kk]) = pa.u;
            *reinterpret_cast<uint2*>(&Blds[row * 40 + kk]) = pb.u;
        }
        __syncthreads();

        v16bf af[2], bf[2];
        #pragma unroll
        for (int i = 0; i < 2; ++i) {
            int m = wm + 16 * i + l15;
            af[i] = ld_frag(&Alds[m * 40 + 8 * hh], &Alds[m * 40 + 16 + 8 * hh]);
        }
        #pragma unroll
        for (int j = 0; j < 2; ++j) {
            int n = wn + 16 * j + l15;
            bf[j] = ld_frag(&Blds[n * 40 + 8 * hh], &Blds[n * 40 + 16 + 8 * hh]);
        }
        #pragma unroll
        for (int i = 0; i < 2; ++i)
            #pragma unroll
            for (int j = 0; j < 2; ++j)
                acc[i][j] = wmma_bf16(af[i], bf[j], acc[i][j]);
    }

    #pragma unroll
    for (int i = 0; i < 2; ++i)
        #pragma unroll
        for (int j = 0; j < 2; ++j)
            #pragma unroll
            for (int r = 0; r < 8; ++r) {
                int gm = M0 + wm + 16 * i + r + 8 * hh;
                int gn = N0 + wn + 16 * j + l15;
                int b = gm >> 11, t = gm & (SEQ - 1);
                int head = gn >> 6, dk = gn & (DKH - 1);
                size_t idx = (((size_t)(b * NHEADS + head) * SEQ) + t) * DKH + dk;
                Out[idx] = (__bf16)acc[i][j][r];
            }
}

// ---------------------------------------------------------------------------
// Kernel 2: flash attention. K tile fetched by the Tensor Data Mover (one
// 16-row padded 2-D DMA per wave, TENSORcnt-tracked); V transposed manually.
// grid = (B*NH, SEQ/64); block = 128 (4 waves x 16 query rows).
// ---------------------------------------------------------------------------
__global__ __launch_bounds__(128)
void flash_attn_kernel(const __bf16* __restrict__ Qg,
                       const __bf16* __restrict__ Kg,
                       const __bf16* __restrict__ Vg,
                       __bf16* __restrict__ Yg)
{
    __shared__ __align__(16) __bf16 Klds[64 * 72];      // [s][dk], 144B stride
    __shared__ __align__(16) __bf16 Vlds[64 * 72];      // transposed [dk][s]
    __shared__ __align__(16) __bf16 Plds[4][16 * 72];   // per-wave P tile

    const int tid  = threadIdx.x;
    const int wave = tid >> 5, lane = tid & 31;
    const int l15  = lane & 15, hh = lane >> 4;

    const int bh = blockIdx.x;
    const int t0 = blockIdx.y * 64;

    const size_t base = (size_t)bh * SEQ * DKH;
    const __bf16* Q = Qg + base;
    const __bf16* K = Kg + base;
    const __bf16* V = Vg + base;

    const unsigned int klds_base = lds_off32(Klds) + (unsigned)wave * 16u * 144u;

    v16bf qf[2];
    {
        const __bf16* qr = Q + (size_t)(t0 + 16 * wave + l15) * DKH;
        qf[0] = ld_frag(qr + 8 * hh,      qr + 16 + 8 * hh);
        qf[1] = ld_frag(qr + 32 + 8 * hh, qr + 48 + 8 * hh);
    }

    float mrow[8], lrow[8];
    v8f o[4];
    #pragma unroll
    for (int r = 0; r < 8; ++r) { mrow[r] = -3.0e38f; lrow[r] = 0.0f; }
    #pragma unroll
    for (int f = 0; f < 4; ++f) o[f] = zero8();

    const float scale = 0.125f;  // 1/sqrt(64)

    for (int s0 = 0; s0 < SEQ; s0 += 64) {
        __syncthreads();
        // K tile: TDM DMA, 16 rows per wave, row pad 64->72 bf16 (pad_iv=4)
        tdm_load_tile_bf16(K + (size_t)(s0 + 16 * wave) * DKH, klds_base,
                           /*rows=*/16, /*width=*/DKH,
                           /*row_stride=*/DKH, /*tensor_rows=*/SEQ,
                           /*pad_iv=*/4);
        // V tile: manual transposed store into [dk][s]
        #pragma unroll
        for (int i = 0; i < 4; ++i) {
            int c    = tid + 128 * i;
            int row  = c >> 3;
            int col8 = (c & 7) * 8;
            uint4 vv = *reinterpret_cast<const uint4*>(V + (size_t)(s0 + row) * DKH + col8);
            const __bf16* vp = reinterpret_cast<const __bf16*>(&vv);
            #pragma unroll
            for (int j = 0; j < 8; ++j)
                Vlds[(col8 + j) * 72 + row] = vp[j];
        }
        __builtin_amdgcn_s_wait_tensorcnt(0);   // own DMA landed in LDS
        __syncthreads();                        // publish K+V tiles

        // S = Q @ K^T
        v8f s[4];
        #pragma unroll
        for (int fn = 0; fn < 4; ++fn) {
            int n = 16 * fn + l15;
            v16bf b0 = ld_frag(&Klds[n * 72 + 8 * hh],      &Klds[n * 72 + 16 + 8 * hh]);
            v16bf b1 = ld_frag(&Klds[n * 72 + 32 + 8 * hh], &Klds[n * 72 + 48 + 8 * hh]);
            v8f c = zero8();
            c = wmma_bf16(qf[0], b0, c);
            c = wmma_bf16(qf[1], b1, c);
            s[fn] = c;
        }

        // Online softmax
        float alpha[8], rsum[8];
        #pragma unroll
        for (int r = 0; r < 8; ++r) {
            float v = fmaxf(fmaxf(s[0][r], s[1][r]), fmaxf(s[2][r], s[3][r])) * scale;
            #pragma unroll
            for (int msk = 1; msk < 16; msk <<= 1)
                v = fmaxf(v, __shfl_xor(v, msk, 32));
            float mn = fmaxf(mrow[r], v);
            alpha[r] = __expf(mrow[r] - mn);
            mrow[r]  = mn;
            rsum[r]  = 0.0f;
        }
        #pragma unroll
        for (int fn = 0; fn < 4; ++fn)
            #pragma unroll
            for (int r = 0; r < 8; ++r) {
                float p = __expf(s[fn][r] * scale - mrow[r]);
                rsum[r] += p;
                Plds[wave][(r + 8 * hh) * 72 + 16 * fn + l15] = (__bf16)p;
            }
        #pragma unroll
        for (int r = 0; r < 8; ++r) {
            float v = rsum[r];
            #pragma unroll
            for (int msk = 1; msk < 16; msk <<= 1)
                v += __shfl_xor(v, msk, 32);
            lrow[r] = lrow[r] * alpha[r] + v;
            #pragma unroll
            for (int f = 0; f < 4; ++f) o[f][r] *= alpha[r];
        }

        asm volatile("s_wait_dscnt 0" ::: "memory");  // wave-local P RAW

        // O += P @ V
        v16bf pa[2];
        {
            const __bf16* pr = &Plds[wave][l15 * 72];
            pa[0] = ld_frag(pr + 8 * hh,      pr + 16 + 8 * hh);
            pa[1] = ld_frag(pr + 32 + 8 * hh, pr + 48 + 8 * hh);
        }
        #pragma unroll
        for (int fn = 0; fn < 4; ++fn) {
            int n = 16 * fn + l15;
            v16bf b0 = ld_frag(&Vlds[n * 72 + 8 * hh],      &Vlds[n * 72 + 16 + 8 * hh]);
            v16bf b1 = ld_frag(&Vlds[n * 72 + 32 + 8 * hh], &Vlds[n * 72 + 48 + 8 * hh]);
            o[fn] = wmma_bf16(pa[0], b0, o[fn]);
            o[fn] = wmma_bf16(pa[1], b1, o[fn]);
        }
    }

    const int b = bh >> 4, head = bh & 15;
    #pragma unroll
    for (int r = 0; r < 8; ++r) {
        float inv = 1.0f / lrow[r];
        int t = t0 + 16 * wave + r + 8 * hh;
        size_t rowbase = ((size_t)b * SEQ + t) * D_MODEL + head * DKH;
        #pragma unroll
        for (int fn = 0; fn < 4; ++fn)
            Yg[rowbase + 16 * fn + l15] = (__bf16)(o[fn][r] * inv);
    }
}

// ---------------------------------------------------------------------------
// Kernel 3: output projection. A tile (bf16) fetched via TDM; B converted
// fp32->bf16 manually. grid = (N/64, M/64); block = 128.
// ---------------------------------------------------------------------------
__global__ __launch_bounds__(128)
void oproj_gemm_kernel(const __bf16* __restrict__ Y,
                       const float* __restrict__ Wo,
                       float* __restrict__ Outp)
{
    __shared__ __align__(16) __bf16 Alds[64 * 40];   // 80B row stride
    __shared__ __align__(16) __bf16 Blds[64 * 40];

    const int tid  = threadIdx.x;
    const int wave = tid >> 5, lane = tid & 31;
    const int l15  = lane & 15, hh = lane >> 4;
    const int wm = (wave >> 1) * 32, wn = (wave & 1) * 32;
    const int M0 = blockIdx.y * 64,  N0 = blockIdx.x * 64;

    const unsigned int alds_base = lds_off32(Alds) + (unsigned)wave * 16u * 80u;

    v8f acc[2][2];
    #pragma unroll
    for (int i = 0; i < 2; ++i)
        #pragma unroll
        for (int j = 0; j < 2; ++j) acc[i][j] = zero8();

    for (int k0 = 0; k0 < D_MODEL; k0 += 32) {
        __syncthreads();
        // A tile: TDM, 16 rows/wave of 32 bf16, pad 32->40 bf16 (pad_iv=3)
        tdm_load_tile_bf16(Y + (size_t)(M0 + 16 * wave) * D_MODEL + k0, alds_base,
                           /*rows=*/16, /*width=*/32,
                           /*row_stride=*/D_MODEL, /*tensor_rows=*/MROWS,
                           /*pad_iv=*/3);
        #pragma unroll
        for (int i = 0; i < 4; ++i) {
            int c   = tid + 128 * i;
            int row = c >> 3;
            int kk  = (c & 7) * 4;
            float4 b = *reinterpret_cast<const float4*>(Wo + (size_t)(N0 + row) * D_MODEL + k0 + kk);
            union { __bf16 h[4]; uint2 u; } pb;
            pb.h[0] = (__bf16)b.x; pb.h[1] = (__bf16)b.y; pb.h[2] = (__bf16)b.z; pb.h[3] = (__bf16)b.w;
            *reinterpret_cast<uint2*>(&Blds[row * 40 + kk]) = pb.u;
        }
        __builtin_amdgcn_s_wait_tensorcnt(0);
        __syncthreads();

        v16bf af[2], bf[2];
        #pragma unroll
        for (int i = 0; i < 2; ++i) {
            int m = wm + 16 * i + l15;
            af[i] = ld_frag(&Alds[m * 40 + 8 * hh], &Alds[m * 40 + 16 + 8 * hh]);
        }
        #pragma unroll
        for (int j = 0; j < 2; ++j) {
            int n = wn + 16 * j + l15;
            bf[j] = ld_frag(&Blds[n * 40 + 8 * hh], &Blds[n * 40 + 16 + 8 * hh]);
        }
        #pragma unroll
        for (int i = 0; i < 2; ++i)
            #pragma unroll
            for (int j = 0; j < 2; ++j)
                acc[i][j] = wmma_bf16(af[i], bf[j], acc[i][j]);
    }

    #pragma unroll
    for (int i = 0; i < 2; ++i)
        #pragma unroll
        for (int j = 0; j < 2; ++j)
            #pragma unroll
            for (int r = 0; r < 8; ++r) {
                int gm = M0 + wm + 16 * i + r + 8 * hh;
                int gn = N0 + wn + 16 * j + l15;
                Outp[(size_t)gm * D_MODEL + gn] = acc[i][j][r];
            }
}

// ---------------------------------------------------------------------------
extern "C" void kernel_launch(void* const* d_in, const int* in_sizes, int n_in,
                              void* d_out, int out_size, void* d_ws, size_t ws_size,
                              hipStream_t stream) {
    (void)in_sizes; (void)n_in; (void)out_size; (void)ws_size;

    const float* X  = (const float*)d_in[0];
    const float* Wq = (const float*)d_in[1];
    const float* Wk = (const float*)d_in[2];
    const float* Wv = (const float*)d_in[3];
    const float* Wo = (const float*)d_in[4];
    float* Out = (float*)d_out;

    __bf16* ws = (__bf16*)d_ws;
    const size_t SZ = (size_t)MROWS * D_MODEL;
    __bf16* Qw = ws;
    __bf16* Kw = ws + SZ;
    __bf16* Vw = ws + 2 * SZ;
    __bf16* Yw = ws + 3 * SZ;

    dim3 g1(D_MODEL / 64, MROWS / 64, 3);
    qkv_gemm_kernel<<<g1, 128, 0, stream>>>(X, Wq, Wk, Wv, ws);

    dim3 g2(BATCH * NHEADS, SEQ / 64);
    flash_attn_kernel<<<g2, 128, 0, stream>>>(Qw, Kw, Vw, Yw);

    dim3 g3(D_MODEL / 64, MROWS / 64);
    oproj_gemm_kernel<<<g3, 128, 0, stream>>>(Yw, Wo, Out);
}
